// InterAttention_77936476554003
// MI455X (gfx1250) — matrix operand order repaired and checked by reading, not compiled, not run on hardware
//
#include <hip/hip_runtime.h>
#include <hip/hip_bf16.h>

// ---------------------------------------------------------------------------
// Problem constants
// ---------------------------------------------------------------------------
#define DIM   1024
#define NL1   256
#define NL2   320
#define NB    64
#define M1R   (NL1 * NB)         // 16384 rows from r1
#define M2R   (NL2 * NB)         // 20480 rows from r2
#define MTOT  (M1R + M2R)        // 36864 rows total

typedef __attribute__((ext_vector_type(16))) __bf16 v16bf;
typedef __attribute__((ext_vector_type(8)))  __bf16 v8bf;
typedef __attribute__((ext_vector_type(8)))  float  v8f;

union V16U { v16bf v; v8bf h[2]; };

// ---------------------------------------------------------------------------
// Helpers
// ---------------------------------------------------------------------------
__device__ __forceinline__ unsigned short f2bf(float f) {
    unsigned u = __float_as_uint(f);
    u += 0x7fffu + ((u >> 16) & 1u);   // round-to-nearest-even
    return (unsigned short)(u >> 16);
}

// CDNA5 async copy: 16B from global -> LDS, tracked by ASYNCcnt.
// VDST carries the per-lane LDS byte offset, VADDR the 64-bit global address.
__device__ __forceinline__ void async_cp16(unsigned ldsOff, const void* g) {
    asm volatile("global_load_async_to_lds_b128 %0, %1, off"
                 :: "v"(ldsOff), "v"((unsigned long long)g)
                 : "memory");
}
__device__ __forceinline__ void wait_async4() {
#if __has_builtin(__builtin_amdgcn_s_wait_asynccnt)
    __builtin_amdgcn_s_wait_asynccnt(4);
#else
    asm volatile("s_wait_asynccnt 4" ::: "memory");
#endif
}
__device__ __forceinline__ void wait_async0() {
#if __has_builtin(__builtin_amdgcn_s_wait_asynccnt)
    __builtin_amdgcn_s_wait_asynccnt(0);
#else
    asm volatile("s_wait_asynccnt 0" ::: "memory");
#endif
}
__device__ __forceinline__ unsigned lds_off_u32(const void* p) {
    // generic pointer to LDS: low 32 bits are the LDS byte offset
    return (unsigned)(unsigned long long)p;
}

// --- fragments straight from global (used by the scores GEMM) --------------
__device__ __forceinline__ v16bf load_frag_a(const unsigned short* __restrict__ base,
                                             int ld, int mBase, int k, int lane) {
    const int h = lane >> 4, r = lane & 15;
    const unsigned short* p = base + (size_t)(mBase + r) * ld + k + 8 * h;
    V16U u;
    u.h[0] = *(const v8bf*)(p);
    u.h[1] = *(const v8bf*)(p + 16);
    return u.v;
}
__device__ __forceinline__ v16bf load_frag_b(const unsigned short* __restrict__ base,
                                             int ld, int nBase, int k, int lane) {
    const int h = lane >> 4, c = lane & 15;
    const unsigned short* p = base + (size_t)(nBase + c) * ld + k + 8 * h;
    V16U u;
    u.h[0] = *(const v8bf*)(p);
    u.h[1] = *(const v8bf*)(p + 16);
    return u.v;
}

// --- fragments from padded LDS tiles (row stride 40 elems = 80B = 20 banks) -
#define LDT 40
__device__ __forceinline__ v16bf lds_frag_a(const unsigned short* base, int row, int lane) {
    const int h = lane >> 4;
    const unsigned short* p = base + (size_t)(row + (lane & 15)) * LDT + 8 * h;
    V16U u;
    u.h[0] = *(const v8bf*)(p);
    u.h[1] = *(const v8bf*)(p + 16);
    return u.v;
}
__device__ __forceinline__ v16bf lds_frag_b(const unsigned short* base, int row, int lane) {
    const int h = lane >> 4;
    const unsigned short* p = base + (size_t)(row + (lane & 15)) * LDT + 8 * h;
    V16U u;
    u.h[0] = *(const v8bf*)(p);
    u.h[1] = *(const v8bf*)(p + 16);
    return u.v;
}

__device__ __forceinline__ float wave_max(float v) {
    #pragma unroll
    for (int m = 16; m > 0; m >>= 1) v = fmaxf(v, __shfl_xor(v, m, 32));
    return v;
}
__device__ __forceinline__ float wave_sum(float v) {
    #pragma unroll
    for (int m = 16; m > 0; m >>= 1) v += __shfl_xor(v, m, 32);
    return v;
}

// ---------------------------------------------------------------------------
// Core: 128x128 workgroup tile, K=1024, bf16 WMMA, async-LDS TRIPLE buffer.
// 256 threads = 8 waves (2M x 4N); wave tile 64x32 (4x2 subtiles).
// Two stages of async copies are kept in flight (4 ops/thread/stage), so
// before computing stage s we only need s_wait_asynccnt <= 4 (in-order
// completion => oldest stage has landed), then a workgroup barrier.
// ---------------------------------------------------------------------------
template <class Epi>
__device__ __forceinline__ void gemm_lds_core(const unsigned short* __restrict__ A,
                                              const unsigned short* __restrict__ W,
                                              int mBlock, int nBlock, Epi&& epi) {
    constexpr int K = DIM;
    constexpr int NSTAGE = K / 32;
    __shared__ unsigned short lsA[3][128 * LDT];   // 3 x 10KB
    __shared__ unsigned short lsB[3][128 * LDT];   // 3 x 10KB  (60KB total)

    const int tid  = threadIdx.x;
    const int lane = tid & 31, wave = tid >> 5;
    const int mW = (wave >> 2) * 64;     // wave-local tile origin in M
    const int nW = (wave & 3) * 32;      // wave-local tile origin in N

    // staging map: chunk c = tid + 256*q (q=0,1); row = c>>2, 16B chunk (c&3)
    const unsigned short* gAq[2];
    const unsigned short* gBq[2];
    unsigned dAq[3][2], dBq[3][2];
    #pragma unroll
    for (int q = 0; q < 2; ++q) {
        int c = tid + 256 * q;
        int row = c >> 2, cc = (c & 3) * 8;
        gAq[q] = A + (size_t)(mBlock + row) * K + cc;
        gBq[q] = W + (size_t)(nBlock + row) * K + cc;
        #pragma unroll
        for (int b = 0; b < 3; ++b) {
            dAq[b][q] = lds_off_u32(&lsA[b][row * LDT + cc]);
            dBq[b][q] = lds_off_u32(&lsB[b][row * LDT + cc]);
        }
    }

    auto stage = [&](int buf, int k0) {
        #pragma unroll
        for (int q = 0; q < 2; ++q) {
            async_cp16(dAq[buf][q], gAq[q] + k0);
            async_cp16(dBq[buf][q], gBq[q] + k0);
        }
    };

    stage(0, 0);
    stage(1, 32);

    v8f acc[4][2] = {};
    int bufC = 0;          // buffer computed this stage
    int bufI = 2;          // buffer to fill with stage s+2
    for (int s = 0; s < NSTAGE; ++s) {
        if (s < NSTAGE - 1) wait_async4();
        else                wait_async0();
        __syncthreads();
        if (s + 2 < NSTAGE) stage(bufI, (s + 2) * 32);

        // preload all fragments, then burst the 8 WMMAs
        v16bf b0 = lds_frag_b(lsB[bufC], nW + 0,  lane);
        v16bf b1 = lds_frag_b(lsB[bufC], nW + 16, lane);
        v16bf afr[4];
        #pragma unroll
        for (int ms = 0; ms < 4; ++ms)
            afr[ms] = lds_frag_a(lsA[bufC], mW + 16 * ms, lane);
        #pragma unroll
        for (int ms = 0; ms < 4; ++ms) {
            acc[ms][0] = __builtin_amdgcn_wmma_f32_16x16x32_bf16(
                false, afr[ms], false, b0, (short)0, acc[ms][0], false, false);
            acc[ms][1] = __builtin_amdgcn_wmma_f32_16x16x32_bf16(
                false, afr[ms], false, b1, (short)0, acc[ms][1], false, false);
        }

        bufC = (bufC == 2) ? 0 : bufC + 1;
        bufI = (bufI == 2) ? 0 : bufI + 1;
    }

    const int h = lane >> 4, c = lane & 15;
    #pragma unroll
    for (int ms = 0; ms < 4; ++ms)
        #pragma unroll
        for (int ns = 0; ns < 2; ++ns) {
            int n = nBlock + nW + ns * 16 + c;
            #pragma unroll
            for (int r = 0; r < 8; ++r) {
                int m = mBlock + mW + ms * 16 + h * 8 + r;
                epi(m, n, acc[ms][ns][r]);
            }
        }
}

// ---------------------------------------------------------------------------
// K1: f32 -> bf16 conversion (4 elements / thread)
// ---------------------------------------------------------------------------
__global__ void k_cvt_bf16(const float* __restrict__ src,
                           unsigned short* __restrict__ dst, int n) {
    int i4 = (blockIdx.x * blockDim.x + threadIdx.x) * 4;
    if (i4 < n) {
        float4 f = *(const float4*)(src + i4);
        unsigned long long pk =
            (unsigned long long)f2bf(f.x)
          | ((unsigned long long)f2bf(f.y) << 16)
          | ((unsigned long long)f2bf(f.z) << 32)
          | ((unsigned long long)f2bf(f.w) << 48);
        *(unsigned long long*)(dst + i4) = pk;
    }
}

// ---------------------------------------------------------------------------
// K2: H = relu(X @ W1^T + b1), bf16 in, bf16 out.
// ---------------------------------------------------------------------------
__global__ void __launch_bounds__(256)
k_gemm_relu(const unsigned short* __restrict__ A,
            const unsigned short* __restrict__ W,
            const float* __restrict__ bias,
            unsigned short* __restrict__ C) {
    gemm_lds_core(A, W, blockIdx.y * 128, blockIdx.x * 128,
        [&](int m, int n, float v) {
            v += bias[n];
            v = v > 0.0f ? v : 0.0f;
            C[(size_t)m * DIM + n] = f2bf(v);
        });
}

// ---------------------------------------------------------------------------
// K3: Y = H @ W2^T + b2, scattered from (L*B) row order to (B,L,D) layout;
// stores f32 (for exact pooling) and bf16 (for the scores WMMA GEMM).
// ---------------------------------------------------------------------------
__global__ void __launch_bounds__(256)
k_gemm_scatter(const unsigned short* __restrict__ A,
               const unsigned short* __restrict__ W,
               const float* __restrict__ bias,
               float* __restrict__ Y1f, float* __restrict__ Y2f,
               unsigned short* __restrict__ Y1b, unsigned short* __restrict__ Y2b) {
    gemm_lds_core(A, W, blockIdx.y * 128, blockIdx.x * 128,
        [&](int m, int n, float v) {
            v += bias[n];
            if (m < M1R) {
                int l = m >> 6, bb = m & 63;              // row = l*64 + b
                size_t o = ((size_t)bb * NL1 + l) * DIM + n;
                Y1f[o] = v; Y1b[o] = f2bf(v);
            } else {
                int m2 = m - M1R;
                int l = m2 >> 6, bb = m2 & 63;
                size_t o = ((size_t)bb * NL2 + l) * DIM + n;
                Y2f[o] = v; Y2b[o] = f2bf(v);
            }
        });
}

// ---------------------------------------------------------------------------
// K4: per-batch scores S[b] = Y1[b] @ Y2[b]^T  (256 x 320, K = 1024).
// 128 threads = 4 waves (2M x 2N). Block tile 128x64, wave tile 64x32.
// Direct global fragment loads (Y tiles are L2-hot), with prefetch.
// ---------------------------------------------------------------------------
__global__ void __launch_bounds__(128)
k_scores(const unsigned short* __restrict__ Y1b,
         const unsigned short* __restrict__ Y2b,
         float* __restrict__ S) {
    const int K = DIM;
    const int b = blockIdx.z;
    const unsigned short* A  = Y1b + (size_t)b * NL1 * DIM;
    const unsigned short* Wm = Y2b + (size_t)b * NL2 * DIM;
    float* Sb = S + (size_t)b * NL1 * NL2;

    const int lane = threadIdx.x & 31, wave = threadIdx.x >> 5;
    const int mWave = blockIdx.y * 128 + (wave >> 1) * 64;
    const int nWave = blockIdx.x * 64  + (wave & 1) * 32;

    v8f acc[4][2] = {};
    for (int k = 0; k < K; k += 32) {
        if (k + 128 < K) {
            __builtin_prefetch(A  + (size_t)(mWave + (lane & 15)) * K + k + 128);
            __builtin_prefetch(Wm + (size_t)(nWave + (lane & 15)) * K + k + 128);
        }
        v16bf b0 = load_frag_b(Wm, K, nWave +  0, k, lane);
        v16bf b1 = load_frag_b(Wm, K, nWave + 16, k, lane);
        #pragma unroll
        for (int ms = 0; ms < 4; ++ms) {
            v16bf a = load_frag_a(A, K, mWave + 16 * ms, k, lane);
            acc[ms][0] = __builtin_amdgcn_wmma_f32_16x16x32_bf16(
                false, a, false, b0, (short)0, acc[ms][0], false, false);
            acc[ms][1] = __builtin_amdgcn_wmma_f32_16x16x32_bf16(
                false, a, false, b1, (short)0, acc[ms][1], false, false);
        }
    }

    const int h = lane >> 4, c = lane & 15;
    #pragma unroll
    for (int ms = 0; ms < 4; ++ms)
        #pragma unroll
        for (int ns = 0; ns < 2; ++ns) {
            int n = nWave + ns * 16 + c;
            #pragma unroll
            for (int r = 0; r < 8; ++r) {
                int m = mWave + ms * 16 + h * 8 + r;
                Sb[(size_t)m * NL2 + n] = acc[ms][ns][r];
            }
        }
}

// ---------------------------------------------------------------------------
// K5a: row stats (softmax over j): one wave per (b,i). 320/32 = 10 elems/lane.
// ---------------------------------------------------------------------------
__global__ void __launch_bounds__(256)
k_rowstats(const float* __restrict__ S, float* __restrict__ rmax, float* __restrict__ rsum) {
    int waveId = blockIdx.x * 8 + (threadIdx.x >> 5);
    int lane = threadIdx.x & 31;
    int b = waveId / NL1, i = waveId % NL1;
    const float* row = S + ((size_t)b * NL1 + i) * NL2;
    float v[10];
    float m = -3.0e38f;
    #pragma unroll
    for (int t = 0; t < 10; ++t) { v[t] = row[lane + 32 * t]; m = fmaxf(m, v[t]); }
    m = wave_max(m);
    float s = 0.0f;
    #pragma unroll
    for (int t = 0; t < 10; ++t) s += __expf(v[t] - m);
    s = wave_sum(s);
    if (lane == 0) { rmax[waveId] = m; rsum[waveId] = s; }
}

// ---------------------------------------------------------------------------
// K5b: column stats + w1[j] = sum_i softmax_row(S)[i,j]. One wave per (b,j).
// ---------------------------------------------------------------------------
__global__ void __launch_bounds__(256)
k_colstats_w1(const float* __restrict__ S,
              const float* __restrict__ rmax, const float* __restrict__ rsum,
              float* __restrict__ cmax, float* __restrict__ csum,
              float* __restrict__ w1) {
    int waveId = blockIdx.x * 8 + (threadIdx.x >> 5);
    int lane = threadIdx.x & 31;
    int b = waveId / NL2, j = waveId % NL2;
    const float* col = S + (size_t)b * NL1 * NL2 + j;
    float v[8];
    float m = -3.0e38f;
    #pragma unroll
    for (int t = 0; t < 8; ++t) { v[t] = col[(size_t)(lane + 32 * t) * NL2]; m = fmaxf(m, v[t]); }
    m = wave_max(m);
    float s = 0.0f, w = 0.0f;
    #pragma unroll
    for (int t = 0; t < 8; ++t) {
        int i = lane + 32 * t;
        s += __expf(v[t] - m);
        w += __expf(v[t] - rmax[b * NL1 + i]) / rsum[b * NL1 + i];
    }
    s = wave_sum(s); w = wave_sum(w);
    if (lane == 0) { cmax[waveId] = m; csum[waveId] = s; w1[waveId] = w; }
}

// ---------------------------------------------------------------------------
// K5c: w2[i] = sum_j softmax_col(S)[i,j]. One wave per (b,i).
// ---------------------------------------------------------------------------
__global__ void __launch_bounds__(256)
k_w2(const float* __restrict__ S,
     const float* __restrict__ cmax, const float* __restrict__ csum,
     float* __restrict__ w2) {
    int waveId = blockIdx.x * 8 + (threadIdx.x >> 5);
    int lane = threadIdx.x & 31;
    int b = waveId / NL1, i = waveId % NL1;
    const float* row = S + ((size_t)b * NL1 + i) * NL2;
    float w = 0.0f;
    #pragma unroll
    for (int t = 0; t < 10; ++t) {
        int j = lane + 32 * t;
        w += __expf(row[j] - cmax[b * NL2 + j]) / csum[b * NL2 + j];
    }
    w = wave_sum(w);
    if (lane == 0) w2[waveId] = w;
}

// ---------------------------------------------------------------------------
// K6: pools. side 0: sweep Y1f -> r1_pool[:D] and r2_pool[D:]
//            side 1: sweep Y2f -> r2_pool[:D] and r1_pool[D:]
// out layout: [r1_pool (64 x 2048), r2_pool (64 x 2048)] flat.
// ---------------------------------------------------------------------------
__global__ void __launch_bounds__(256)
k_pool(const float* __restrict__ Y1f, const float* __restrict__ Y2f,
       const float* __restrict__ w1, const float* __restrict__ w2,
       float* __restrict__ out) {
    int d = blockIdx.x * 256 + threadIdx.x;
    int b = blockIdx.y;
    float acc0 = 0.0f, acc1 = 0.0f;
    if (blockIdx.z == 0) {
        const float* Y = Y1f + (size_t)b * NL1 * DIM + d;
        for (int i = 0; i < NL1; ++i) {
            float y = Y[(size_t)i * DIM];
            acc0 += y;
            acc1 += w2[b * NL1 + i] * y;
        }
        out[(size_t)b * 2048 + d]                            = acc0 * (1.0f / NL1);
        out[(size_t)NB * 2048 + (size_t)b * 2048 + 1024 + d] = acc1 * (1.0f / NL2);
    } else {
        const float* Y = Y2f + (size_t)b * NL2 * DIM + d;
        for (int j = 0; j < NL2; ++j) {
            float y = Y[(size_t)j * DIM];
            acc0 += y;
            acc1 += w1[b * NL2 + j] * y;
        }
        out[(size_t)NB * 2048 + (size_t)b * 2048 + d] = acc0 * (1.0f / NL2);
        out[(size_t)b * 2048 + 1024 + d]              = acc1 * (1.0f / NL1);
    }
}

// ---------------------------------------------------------------------------
// Host launcher
// ---------------------------------------------------------------------------
extern "C" void kernel_launch(void* const* d_in, const int* in_sizes, int n_in,
                              void* d_out, int out_size, void* d_ws, size_t ws_size,
                              hipStream_t stream) {
    const float* r1 = (const float*)d_in[0];
    const float* r2 = (const float*)d_in[1];
    const float* W1 = (const float*)d_in[2];
    const float* b1 = (const float*)d_in[3];
    const float* W2 = (const float*)d_in[4];
    const float* b2 = (const float*)d_in[5];
    float* out = (float*)d_out;

    char* w = (char*)d_ws;
    size_t off = 0;
    auto take = [&](size_t bytes) -> void* {
        void* p = w + off;
        off = (off + bytes + 255) & ~(size_t)255;
        return p;
    };
    unsigned short* Xb  = (unsigned short*)take((size_t)MTOT * DIM * 2);
    unsigned short* W1b = (unsigned short*)take((size_t)DIM * DIM * 2);
    unsigned short* W2b = (unsigned short*)take((size_t)DIM * DIM * 2);
    unsigned short* Hb  = (unsigned short*)take((size_t)MTOT * DIM * 2);
    float*          Y1f = (float*)take((size_t)NB * NL1 * DIM * 4);
    float*          Y2f = (float*)take((size_t)NB * NL2 * DIM * 4);
    unsigned short* Y1b = (unsigned short*)take((size_t)NB * NL1 * DIM * 2);
    unsigned short* Y2b = (unsigned short*)take((size_t)NB * NL2 * DIM * 2);
    float*          S   = (float*)take((size_t)NB * NL1 * NL2 * 4);
    float*          rmaxb = (float*)take((size_t)NB * NL1 * 4);
    float*          rsumb = (float*)take((size_t)NB * NL1 * 4);
    float*          cmaxb = (float*)take((size_t)NB * NL2 * 4);
    float*          csumb = (float*)take((size_t)NB * NL2 * 4);
    float*          w1v   = (float*)take((size_t)NB * NL2 * 4);
    float*          w2v   = (float*)take((size_t)NB * NL1 * 4);
    (void)ws_size; (void)in_sizes; (void)n_in; (void)out_size;

    // K1: convert inputs to bf16
    {
        int n;
        n = M1R * DIM;
        k_cvt_bf16<<<(n / 4 + 255) / 256, 256, 0, stream>>>(r1, Xb, n);
        n = M2R * DIM;
        k_cvt_bf16<<<(n / 4 + 255) / 256, 256, 0, stream>>>(r2, Xb + (size_t)M1R * DIM, n);
        n = DIM * DIM;
        k_cvt_bf16<<<(n / 4 + 255) / 256, 256, 0, stream>>>(W1, W1b, n);
        k_cvt_bf16<<<(n / 4 + 255) / 256, 256, 0, stream>>>(W2, W2b, n);
    }

    // K2: H = relu(X @ W1^T + b1)
    k_gemm_relu<<<dim3(DIM / 128, MTOT / 128), 256, 0, stream>>>(Xb, W1b, b1, Hb);

    // K3: Y = H @ W2^T + b2, scattered to (B, L, D), f32 + bf16
    k_gemm_scatter<<<dim3(DIM / 128, MTOT / 128), 256, 0, stream>>>(
        Hb, W2b, b2, Y1f, Y2f, Y1b, Y2b);

    // K4: per-batch scores
    k_scores<<<dim3(NL2 / 64, NL1 / 128, NB), 128, 0, stream>>>(Y1b, Y2b, S);

    // K5: softmax statistics and attention weight sums
    k_rowstats<<<(NB * NL1) / 8, 256, 0, stream>>>(S, rmaxb, rsumb);
    k_colstats_w1<<<(NB * NL2) / 8, 256, 0, stream>>>(S, rmaxb, rsumb, cmaxb, csumb, w1v);
    k_w2<<<(NB * NL1) / 8, 256, 0, stream>>>(S, cmaxb, csumb, w2v);

    // K6: pools
    k_pool<<<dim3(DIM / 256, NB, 2), 256, 0, stream>>>(Y1f, Y2f, w1v, w2v, out);
}
